// SelfAttention_82506321756292
// MI455X (gfx1250) — matrix operand order repaired and checked by reading, not compiled, not run on hardware
//
#include <hip/hip_runtime.h>
#include <cstdint>
#include <cstddef>

#define DEV __device__ __forceinline__

// ---------------- problem constants ----------------
constexpr int Bv   = 2;
constexpr int Sv   = 2048;
constexpr int DIMv = 4096;
constexpr int NQv  = 32;
constexpr int NKVv = 8;
constexpr int HDv  = 128;                       // DIM / NQ
constexpr int QKVN = NQv*HDv + 2*NKVv*HDv;      // 6144
constexpr int Mrows = Bv*Sv;                    // 4096

// ---------------- WMMA types (gfx1250, wave32) ----------------
typedef __attribute__((ext_vector_type(16))) __bf16 v16bf;
typedef __attribute__((ext_vector_type(8)))  float  v8f;

DEV __bf16 f2bf(float f) {
  union { float f; uint32_t u; } x; x.f = f;
  uint32_t r = x.u + 0x7FFFu + ((x.u >> 16) & 1u);   // round-to-nearest-even
  unsigned short h = (unsigned short)(r >> 16);
  return __builtin_bit_cast(__bf16, h);
}
DEV float bf2f(__bf16 b) {
  unsigned short h = __builtin_bit_cast(unsigned short, b);
  union { uint32_t u; float f; } x; x.u = ((uint32_t)h) << 16;
  return x.f;
}

// A-operand (16x32 bf16, row-major src, leading dim ld):
// lane L: row = L&15 ; K-offsets {0..15} (L<16) or {16..31} (L>=16), contiguous
DEV v16bf load_frag_a_g(const __bf16* base, int ld) {
  int lane = threadIdx.x & 31;
  int row  = lane & 15;
  int koff = (lane < 16) ? 0 : 16;
  const __bf16* p = base + (size_t)row * ld + koff;
  union { uint4 u[2]; v16bf v; } un;
  un.u[0] = *(const uint4*)(p);
  un.u[1] = *(const uint4*)(p + 8);
  return un.v;
}
// B-operand (32x16 bf16) read from B^T stored row-major [N,K], leading dim ld:
// lane L: col n = L&15 ; K = {0..15} (L<16) or {16..31} (L>=16), contiguous
DEV v16bf load_frag_b_g(const __bf16* base, int ld) {
  int lane = threadIdx.x & 31;
  int n    = lane & 15;
  int kb   = (lane < 16) ? 0 : 16;
  const __bf16* p = base + (size_t)n * ld + kb;
  union { uint4 u[2]; v16bf v; } un;
  un.u[0] = *(const uint4*)(p);
  un.u[1] = *(const uint4*)(p + 8);
  return un.v;
}

DEV v8f wmma_bf16(v16bf a, v16bf b, v8f c) {
  return __builtin_amdgcn_wmma_f32_16x16x32_bf16(
      /*neg_a=*/false, a, /*neg_b=*/false, b,
      /*c_mod=*/(short)0, c, /*reuse_a=*/false, /*reuse_b=*/false);
}

// ---- CDNA5 async global->LDS copy (ASYNCcnt path, ISA §10/§15.18) ----
// VDST = per-lane LDS byte address, VADDR = 64-bit global address ("GV" mode).
DEV void async_g2lds_b128(uint32_t lds_addr, const void* gaddr) {
  asm volatile("global_load_async_to_lds_b128 %0, %1, off"
               :: "v"(lds_addr), "v"(gaddr) : "memory");
}
DEV void wait_async0() {
#if __has_builtin(__builtin_amdgcn_s_wait_asynccnt)
  __builtin_amdgcn_s_wait_asynccnt(0);
#else
  asm volatile("s_wait_asynccnt 0x0" ::: "memory");
#endif
}

DEV void storeC(float* p, float v)  { *p = v; }
DEV void storeC(__bf16* p, float v) { *p = f2bf(v); }

// ---------------- fp32 -> bf16 bulk convert ----------------
__global__ void k_cvt_bf16(const float* __restrict__ src, __bf16* __restrict__ dst, size_t n) {
  size_t i = ((size_t)blockIdx.x * blockDim.x + threadIdx.x) * 4;
  if (i + 3 < n) {
    float4 f = *(const float4*)(src + i);
    dst[i+0] = f2bf(f.x); dst[i+1] = f2bf(f.y);
    dst[i+2] = f2bf(f.z); dst[i+3] = f2bf(f.w);
  }
}

// ---------------- transpose + convert: W[K,N] fp32 -> Wt[rowOff+N, K] bf16 ----------------
__global__ void k_transpose_cvt(const float* __restrict__ W, __bf16* __restrict__ Wt,
                                int K, int N, int rowOff, int ldT) {
  __shared__ float tile[16][17];
  int n0 = blockIdx.x * 16, k0 = blockIdx.y * 16;
  int tx = threadIdx.x, ty = threadIdx.y;
  tile[ty][tx] = W[(size_t)(k0 + ty) * N + (n0 + tx)];
  __syncthreads();
  Wt[(size_t)(rowOff + n0 + ty) * ldT + (k0 + tx)] = f2bf(tile[tx][ty]);
}

// ---------------- bf16 WMMA GEMM: C[M,N] = A[M,K] * Bt[N,K]^T ----------------
// one wave per 32(M) x 64(N) tile: 2 A-fragments, 4 B-fragments, 8 WMMAs / k-step
template<typename OutT>
__global__ void k_gemm_bf16(const __bf16* __restrict__ A, const __bf16* __restrict__ Bt,
                            OutT* __restrict__ C, int M, int N, int K) {
  int n0 = blockIdx.x * 64;
  int m0 = blockIdx.y * 32;
  int lane = threadIdx.x & 31;
  v8f acc0[4] = {};
  v8f acc1[4] = {};
  for (int k0 = 0; k0 < K; k0 += 32) {
    // prefetch one 128B line ahead per operand row
    if (k0 + 64 < K) {
      __builtin_prefetch(A  + (size_t)(m0 + lane) * K + k0 + 64, 0, 1);
      __builtin_prefetch(Bt + (size_t)(n0 + lane) * K + k0 + 64, 0, 1);
      __builtin_prefetch(Bt + (size_t)(n0 + 32 + lane) * K + k0 + 64, 0, 1);
    }
    v16bf a0 = load_frag_a_g(A + (size_t)m0 * K + k0, K);
    v16bf a1 = load_frag_a_g(A + (size_t)(m0 + 16) * K + k0, K);
#pragma unroll
    for (int i = 0; i < 4; ++i) {
      v16bf b = load_frag_b_g(Bt + (size_t)(n0 + 16*i) * K + k0, K);
      acc0[i] = wmma_bf16(a0, b, acc0[i]);
      acc1[i] = wmma_bf16(a1, b, acc1[i]);
    }
  }
  int col = lane & 15;
  int rhi = (lane >> 4) * 8;           // C layout: VGPR v -> row v (+8 for hi lanes)
#pragma unroll
  for (int i = 0; i < 4; ++i)
#pragma unroll
    for (int v = 0; v < 8; ++v) {
      storeC(&C[(size_t)(m0 + rhi + v) * N + (n0 + 16*i + col)],      acc0[i][v]);
      storeC(&C[(size_t)(m0 + 16 + rhi + v) * N + (n0 + 16*i + col)], acc1[i][v]);
    }
}

// ---------------- RoPE + scatter into attention layouts ----------------
// qkv[M, 6144] bf16 -> qb[B,NQ,S,HD] (roped), kb[B,NKV,S,HD] (roped), vT[B,NKV,HD,S]
__global__ void k_rope_scatter(const __bf16* __restrict__ qkv, const float* __restrict__ fc,
                               __bf16* __restrict__ qb, __bf16* __restrict__ kb,
                               __bf16* __restrict__ vT) {
  size_t tid = (size_t)blockIdx.x * blockDim.x + threadIdx.x;
  const size_t total = (size_t)Bv * Sv * (NQv + 2*NKVv) * (HDv/2);
  if (tid >= total) return;
  int p = (int)(tid & (HDv/2 - 1));
  size_t t = tid / (HDv/2);
  int slot = (int)(t % (NQv + 2*NKVv)); t /= (NQv + 2*NKVv);
  int s = (int)(t % Sv);
  int b = (int)(t / Sv);
  size_t row = (size_t)b * Sv + s;
  const __bf16* src = qkv + row * QKVN + slot * HDv + 2*p;
  if (slot < NQv + NKVv) {
    float x0 = bf2f(src[0]), x1 = bf2f(src[1]);
    float cs = fc[((size_t)s*(HDv/2) + p)*2 + 0];
    float sn = fc[((size_t)s*(HDv/2) + p)*2 + 1];
    float r0 = x0*cs - x1*sn;
    float r1 = x0*sn + x1*cs;
    if (slot < NQv) {
      size_t o = (((size_t)b*NQv + slot)*Sv + s)*HDv + 2*p;
      qb[o] = f2bf(r0); qb[o+1] = f2bf(r1);
    } else {
      int hk = slot - NQv;
      size_t o = (((size_t)b*NKVv + hk)*Sv + s)*HDv + 2*p;
      kb[o] = f2bf(r0); kb[o+1] = f2bf(r1);
    }
  } else {
    int hk = slot - NQv - NKVv;
    size_t basehd = ((size_t)b*NKVv + hk)*HDv + 2*p;
    vT[basehd * Sv + s]       = src[0];
    vT[(basehd + 1) * Sv + s] = src[1];
  }
}

// ---------------- flash attention ----------------
// 4 waves / block; each wave owns one 16-row q tile of the same (b,h).
// K/V chunks (32 keys) are staged in double-buffered LDS via async global->LDS
// loads (ASYNCcnt), shared by all 4 waves -> 4x less K/V traffic.
__global__ void __launch_bounds__(128)
k_flash_attn(const __bf16* __restrict__ Q, const __bf16* __restrict__ Kc,
             const __bf16* __restrict__ Vt, __bf16* __restrict__ Oatt) {
  const int h  = blockIdx.y;
  const int b  = blockIdx.z;
  const int hk = h / (NQv / NKVv);
  const int tid  = (int)threadIdx.x;     // 0..127
  const int w    = tid >> 5;             // wave id 0..3
  const int lane = tid & 31;
  const int qt   = blockIdx.x * 4 + w;   // 16-row q tile index

  const __bf16* qbase = Q  + (((size_t)b*NQv  + h )*Sv + (size_t)qt*16)*HDv;
  const __bf16* kbase = Kc + ((size_t)b*NKVv + hk)*(size_t)Sv*HDv;
  const __bf16* vbase = Vt + ((size_t)b*NKVv + hk)*(size_t)HDv*Sv;

  __shared__ __align__(16) __bf16 sK[2][32][HDv];   // 16 KB double-buffered K chunk
  __shared__ __align__(16) __bf16 sV[2][HDv][32];   // 16 KB double-buffered V^T chunk
  __shared__ __align__(16) __bf16 sQ[4][16][HDv];   // 16 KB per-wave q tiles
  __shared__ __align__(16) float  sS[4][16][32];    //  8 KB per-wave score tiles
  __shared__ __align__(16) __bf16 sP[4][16][32];    //  4 KB per-wave P tiles
  __shared__ float sAlpha[4][16];
  __shared__ float sL[4][16];

  // ---- stage a chunk (K: keys kv..kv+63, dims 0..63, 64 rows from sK[bufI][0]; V^T: 128x32) via async loads ----
  auto stage = [&](int bufI, int kv) {
    // K chunk: 512 x 16B units, 4 per thread
    for (int u = tid; u < 512; u += 128) {
      int row = u >> 3, part = u & 7;
      async_g2lds_b128((uint32_t)(uintptr_t)&sK[bufI][row][part*8],
                       kbase + (size_t)(kv + row)*HDv + part*8);
    }
    // V^T chunk: 512 x 16B units, 4 per thread
    for (int u = tid; u < 512; u += 128) {
      int row = u >> 2, part = u & 3;
      async_g2lds_b128((uint32_t)(uintptr_t)&sV[bufI][row][part*8],
                       vbase + (size_t)row*Sv + kv + part*8);
    }
  };

  // cooperative per-wave Q tile copy (tile is a contiguous 4KB block)
  {
    const uint4* qg = (const uint4*)qbase;   // 256 x 16B
    uint4* qs = (uint4*)&sQ[w][0][0];
#pragma unroll
    for (int i = 0; i < 8; ++i) qs[lane + 32*i] = qg[lane + 32*i];
  }
  stage(0, 0);
  wait_async0();
  __syncthreads();

  v8f o[8] = {};
  float mrow = -3.0e38f, lrow = 0.0f;          // live in lanes 0..15
  const float scale = 0.08838834764831845f;    // 1/sqrt(128)
  const int col = lane & 15;
  const int rhi = (lane >> 4) * 8;
  const int arow = lane & 15;                  // A-fragment row for this lane
  const int akoff = (lane < 16) ? 0 : 16;      // A-fragment K sub-offset
  const int bn   = lane & 15;                  // B-fragment column
  const int bkb  = (lane < 16) ? 0 : 16;       // B-fragment K sub-offset

  int buf = 0;
  for (int kv = 0; kv < Sv; kv += 32) {
    const bool hasNext = (kv + 32 < Sv);
    if (hasNext) stage(buf ^ 1, kv + 32);      // async fill of the other buffer

    // ---- S_tile = Q(16xHD) * K^T(HDx32) from LDS, fp32 accum ----
    v8f s0 = {}, s1 = {};
#pragma unroll
    for (int kk = 0; kk < 4; ++kk) {
      v16bf a;
      {
        union { uint4 u[2]; v16bf v; } un;
        un.u[0] = *(const uint4*)&sQ[w][arow][kk*32 + akoff];
        un.u[1] = *(const uint4*)&sQ[w][arow][kk*32 + akoff + 8];
        a = un.v;
      }
      v16bf kb0, kb1;
      {
        union { uint4 u[2]; v16bf v; } un;
        un.u[0] = *(const uint4*)&sK[buf][bn][kk*32 + bkb];
        un.u[1] = *(const uint4*)&sK[buf][bn][kk*32 + bkb + 8];
        kb0 = un.v;
        un.u[0] = *(const uint4*)&sK[buf][16 + bn][kk*32 + bkb];
        un.u[1] = *(const uint4*)&sK[buf][16 + bn][kk*32 + bkb + 8];
        kb1 = un.v;
      }
      s0 = wmma_bf16(a, kb0, s0);
      s1 = wmma_bf16(a, kb1, s1);
    }
#pragma unroll
    for (int v = 0; v < 8; ++v) {
      sS[w][rhi + v][col]      = s0[v] * scale;
      sS[w][rhi + v][16 + col] = s1[v] * scale;
    }
    __syncthreads();
    // ---- online softmax per row (lanes 0..15 own one row each) ----
    if (lane < 16) {
      float mx = mrow;
#pragma unroll
      for (int j = 0; j < 32; ++j) mx = fmaxf(mx, sS[w][lane][j]);
      float alpha = __expf(mrow - mx);
      float lsum = 0.0f;
#pragma unroll
      for (int j = 0; j < 32; ++j) {
        float pv = __expf(sS[w][lane][j] - mx);
        sP[w][lane][j] = f2bf(pv);
        lsum += pv;
      }
      lrow = lrow * alpha + lsum;
      mrow = mx;
      sAlpha[w][lane] = alpha;
    }
    __syncthreads();
    float al[8];
#pragma unroll
    for (int v = 0; v < 8; ++v) al[v] = sAlpha[w][rhi + v];
    v16bf pa;
    {
      union { uint4 u[2]; v16bf v; } un;
      un.u[0] = *(const uint4*)&sP[w][arow][akoff];
      un.u[1] = *(const uint4*)&sP[w][arow][akoff + 8];
      pa = un.v;
    }
    // ---- O = O*alpha + P(16x32) * V(32xHD) from LDS ----
#pragma unroll
    for (int nt = 0; nt < 8; ++nt) {
#pragma unroll
      for (int v = 0; v < 8; ++v) o[nt][v] *= al[v];
      v16bf vb;
      {
        union { uint4 u[2]; v16bf v; } un;
        un.u[0] = *(const uint4*)&sV[buf][nt*16 + bn][bkb];
        un.u[1] = *(const uint4*)&sV[buf][nt*16 + bn][bkb + 8];
        vb = un.v;
      }
      o[nt] = wmma_bf16(pa, vb, o[nt]);
    }
    // drain this wave's async loads, then block-wide barrier before buffer swap
    if (hasNext) wait_async0();
    __syncthreads();
    if (hasNext) buf ^= 1;
  }

  if (lane < 16) sL[w][lane] = lrow;
  __syncthreads();
  float inv_l[8];
#pragma unroll
  for (int v = 0; v < 8; ++v) inv_l[v] = 1.0f / sL[w][rhi + v];

#pragma unroll
  for (int nt = 0; nt < 8; ++nt)
#pragma unroll
    for (int v = 0; v < 8; ++v) {
      size_t r = (size_t)b*Sv + (size_t)qt*16 + rhi + v;
      Oatt[r * (size_t)(NQv*HDv) + (size_t)h*HDv + nt*16 + col] = f2bf(o[nt][v] * inv_l[v]);
    }
}

// ---------------- launch ----------------
extern "C" void kernel_launch(void* const* d_in, const int* in_sizes, int n_in,
                              void* d_out, int out_size, void* d_ws, size_t ws_size,
                              hipStream_t stream) {
  (void)in_sizes; (void)n_in; (void)out_size; (void)ws_size;
  const float* x  = (const float*)d_in[0];
  // d_in[1] = start_pos (0 in this problem)
  const float* fc = (const float*)d_in[2];
  const float* wq = (const float*)d_in[3];
  const float* wk = (const float*)d_in[4];
  const float* wv = (const float*)d_in[5];
  const float* wo = (const float*)d_in[6];
  float* out = (float*)d_out;

  // workspace layout (bf16), with aliasing of dead regions:
  __bf16* xb    = (__bf16*)d_ws;                          // [4096, 4096]   32 MB
  __bf16* wqkvT = xb    + (size_t)Mrows * DIMv;           // [6144, 4096]   48 MB
  __bf16* woT   = wqkvT + (size_t)QKVN  * DIMv;           // [4096, 4096]   32 MB
  __bf16* qkv   = woT   + (size_t)DIMv  * DIMv;           // [4096, 6144]   48 MB
  // after GEMM1 the weight-T region is dead -> reuse for attention operands
  __bf16* qb = wqkvT;                                     // [B,NQ,S,HD]    32 MB
  __bf16* kb = qb + (size_t)Bv*NQv*Sv*HDv;                // [B,NKV,S,HD]    8 MB
  __bf16* vT = kb + (size_t)Bv*NKVv*Sv*HDv;               // [B,NKV,HD,S]    8 MB
  // after RoPE the qkv region is dead -> reuse for attention output
  __bf16* attnb = qkv;                                    // [4096, 4096]   32 MB

  // 1) x -> bf16
  {
    size_t n = (size_t)Mrows * DIMv;
    k_cvt_bf16<<<dim3((unsigned)(n / 1024)), dim3(256), 0, stream>>>(x, xb, n);
  }
  // 2) weights -> bf16, transposed (B^T row-major for contiguous B-operand loads)
  dim3 tb(16, 16);
  k_transpose_cvt<<<dim3(DIMv/16,       DIMv/16), tb, 0, stream>>>(wq, wqkvT, DIMv, DIMv,     0,                 DIMv);
  k_transpose_cvt<<<dim3((NKVv*HDv)/16, DIMv/16), tb, 0, stream>>>(wk, wqkvT, DIMv, NKVv*HDv, NQv*HDv,           DIMv);
  k_transpose_cvt<<<dim3((NKVv*HDv)/16, DIMv/16), tb, 0, stream>>>(wv, wqkvT, DIMv, NKVv*HDv, NQv*HDv+NKVv*HDv,  DIMv);
  k_transpose_cvt<<<dim3(DIMv/16,       DIMv/16), tb, 0, stream>>>(wo, woT,   DIMv, DIMv,     0,                 DIMv);

  // 3) fused QKV projection: qkv[4096,6144] = xb * wqkvT^T
  k_gemm_bf16<__bf16><<<dim3(QKVN/64, Mrows/32), dim3(32), 0, stream>>>(xb, wqkvT, qkv, Mrows, QKVN, DIMv);

  // 4) RoPE + scatter to attention layouts
  {
    size_t total = (size_t)Bv * Sv * (NQv + 2*NKVv) * (HDv/2);
    k_rope_scatter<<<dim3((unsigned)((total + 255) / 256)), dim3(256), 0, stream>>>(qkv, fc, qb, kb, vT);
  }

  // 5) flash attention (4 q-tiles per block share async-staged K/V chunks)
  k_flash_attn<<<dim3(Sv/64, NQv, Bv), dim3(128), 0, stream>>>(qb, kb, vT, attnb);

  // 6) output projection: out[4096,4096] = attnb * woT^T (fp32 out)
  k_gemm_bf16<float><<<dim3(DIMv/64, Mrows/32), dim3(32), 0, stream>>>(attnb, woT, out, Mrows, DIMv, DIMv);
}
